// GraphAttentionLayer_90288802496595
// MI455X (gfx1250) — matrix operand order repaired and checked by reading, not compiled, not run on hardware
//
#include <hip/hip_runtime.h>
#include <hip/hip_bf16.h>

typedef _Float16 v16h __attribute__((ext_vector_type(16)));
typedef _Float16 v8h  __attribute__((ext_vector_type(8)));
typedef float    v8f  __attribute__((ext_vector_type(8)));

#define BATCH 16
#define SEQ   2048
#define DIM   256
#define LEAKY 0.2f

// ---------------------------------------------------------------------------
// Kernel 1: W (f32, [k][n]) -> W16T (f16, [n][k])  (transposed for WMMA B reads)
// ---------------------------------------------------------------------------
__global__ void k_prep(const float* __restrict__ W, _Float16* __restrict__ W16T) {
    int idx = blockIdx.x * blockDim.x + threadIdx.x;   // 0..65535
    int n = idx >> 8;
    int k = idx & 255;
    W16T[n * DIM + k] = (_Float16)W[k * DIM + n];
}

// ---------------------------------------------------------------------------
// Kernel 2: Wh = x @ W  via v_wmma_f32_16x16x32_f16.
// W16T (128 KB) is staged ONCE per block in LDS via async-to-LDS copies, so
// the 128-WMMA inner loop feeds from short-latency ds_loads with a 1-deep
// prefetch instead of stalling on global loads. One wave = 16 rows x N=256.
// Stores Wh TRANSPOSED per batch: WhT[b][d][j] (f16) for kernel 6's B reads.
// ---------------------------------------------------------------------------
__global__ void __launch_bounds__(256) k_gemm1(const float* __restrict__ x,
                                               const _Float16* __restrict__ W16T,
                                               _Float16* __restrict__ WhT) {
    __shared__ __align__(64) _Float16 Wlds[DIM * DIM];   // 128 KB

    const int tid  = threadIdx.x;
    const int lane = tid & 31;
    const int wave = tid >> 5;
    const int tile = blockIdx.x * 8 + wave;   // 0..2047 row-tiles
    const int row0 = tile * 16;               // global row (b*SEQ + s)
    const int b    = row0 / SEQ;
    const int s0   = row0 % SEQ;
    const int m    = lane & 15;
    const int hi   = lane >> 4;
    const int kb   = hi * 16;

    // Async-fill LDS with the whole W16T: 8192 x 16B chunks, 32 per thread.
    {
        const unsigned ldsbase = (unsigned)(size_t)(&Wlds[0]);
        unsigned long long g = (unsigned long long)(size_t)W16T;
#pragma unroll
        for (int i = 0; i < 32; ++i) {
            unsigned off = ((unsigned)tid + 256u * (unsigned)i) * 16u;
            asm volatile("global_load_async_to_lds_b128 %0, %1, off"
                         :: "v"(ldsbase + off), "v"(g + off)
                         : "memory");
        }
        asm volatile("s_wait_asynccnt 0x0" ::: "memory");
        __syncthreads();
    }

    // Load A: 16 rows x K=256 of x (read exactly once -> non-temporal),
    // converted f32->f16, 8 K-tiles held in registers.
    v16h A[8];
    const float* xr = x + (size_t)(row0 + m) * DIM;
#pragma unroll
    for (int kt = 0; kt < 8; ++kt) {
        v8f lo = __builtin_nontemporal_load((const v8f*)(xr + kt * 32 + hi * 16));
        v8f hh = __builtin_nontemporal_load((const v8f*)(xr + kt * 32 + hi * 16 + 8));
        v16h a;
#pragma unroll
        for (int e = 0; e < 8; ++e) { a[e] = (_Float16)lo[e]; a[e + 8] = (_Float16)hh[e]; }
        A[kt] = a;
    }

#pragma unroll
    for (int nt = 0; nt < 16; ++nt) {
        v8f acc = {};
        const _Float16* wb = &Wlds[(nt * 16 + m) * DIM + kb];  // N = nt*16 + lane%16
        v16h bm = *(const v16h*)wb;
#pragma unroll
        for (int kt = 0; kt < 8; ++kt) {
            v16h bn = (kt < 7) ? *(const v16h*)(wb + (kt + 1) * 32) : bm;
            acc = __builtin_amdgcn_wmma_f32_16x16x32_f16(
                false, A[kt], false, bm, (short)0, acc, false, false);
            bm = bn;
        }
        // C layout: lane n = lane%16, VGPR v -> M = v + 8*hi. Transposed store:
        // WhT[b][d = nt*16 + n][j = s0 + 8*hi + v] -> 8 contiguous f16 per lane.
        // (WhT is re-read heavily by later kernels -> keep regular-temporal.)
        v8h outv;
#pragma unroll
        for (int v = 0; v < 8; ++v) outv[v] = (_Float16)acc[v];
        _Float16* dst = WhT + ((size_t)(b * DIM + nt * 16 + m) * SEQ) + s0 + hi * 8;
        *(v8h*)dst = outv;
    }
}

// ---------------------------------------------------------------------------
// Kernel 3: s1[r] = Wh[r,:] . a1 ;  s2[r] = Wh[r,:] . a2  (one thread / row)
// ---------------------------------------------------------------------------
__global__ void __launch_bounds__(256) k_s12(const _Float16* __restrict__ WhT,
                                             const float* __restrict__ a,
                                             float* __restrict__ s1,
                                             float* __restrict__ s2) {
    int r = blockIdx.x * blockDim.x + threadIdx.x;   // 0..32767
    int b = r / SEQ, s = r % SEQ;
    const _Float16* base = WhT + (size_t)b * DIM * SEQ + s;
    float acc1 = 0.f, acc2 = 0.f;
    for (int d = 0; d < DIM; ++d) {
        float w = (float)base[(size_t)d * SEQ];
        acc1 += w * a[d];
        acc2 += w * a[DIM + d];
    }
    s1[r] = acc1;
    s2[r] = acc2;
}

// ---------------------------------------------------------------------------
// Kernel 4: m2[b] = max_j s2[b,j]
// ---------------------------------------------------------------------------
__global__ void __launch_bounds__(256) k_m2(const float* __restrict__ s2,
                                            float* __restrict__ m2) {
    __shared__ float red[256];
    int b = blockIdx.x;
    float mx = -3.402823466e38f;
    for (int j = threadIdx.x; j < SEQ; j += 256) mx = fmaxf(mx, s2[b * SEQ + j]);
    red[threadIdx.x] = mx;
    __syncthreads();
    for (int off = 128; off > 0; off >>= 1) {
        if (threadIdx.x < off) red[threadIdx.x] = fmaxf(red[threadIdx.x], red[threadIdx.x + off]);
        __syncthreads();
    }
    if (threadIdx.x == 0) m2[b] = red[0];
}

// ---------------------------------------------------------------------------
// Kernel 5: softmax stats. LeakyReLU is monotone => rowmax = f(s1 + max(s2)).
// ---------------------------------------------------------------------------
__global__ void __launch_bounds__(256) k_denom(const float* __restrict__ s1,
                                               const float* __restrict__ s2,
                                               const float* __restrict__ m2,
                                               float* __restrict__ rowmax,
                                               float* __restrict__ invden) {
    __shared__ float s2s[SEQ];   // 8 KB; whole block is in one batch (256 | 2048)
    int r = blockIdx.x * 256 + threadIdx.x;
    int b = r / SEQ;
    const float* s2b = s2 + b * SEQ;
    for (int j = threadIdx.x; j < SEQ; j += 256) s2s[j] = s2b[j];
    __syncthreads();

    float sv = s1[r];
    float zm = sv + m2[b];
    float rm = zm >= 0.f ? zm : LEAKY * zm;
    float sum = 0.f;
    for (int j = 0; j < SEQ; ++j) {
        float z = sv + s2s[j];
        float l = z >= 0.f ? z : LEAKY * z;
        sum += __expf(l - rm);
    }
    rowmax[r] = rm;
    invden[r] = 1.0f / sum;
}

// ---------------------------------------------------------------------------
// Kernel 6: fused attention + h' = att @ Wh.
// One wave = one 16-row tile x full N=256 (16 WMMA accumulators).
// B-tiles (WhT[b][*][jt*32..+31], 16 KB) are SHARED by all 8 waves of the
// block: double-buffered in LDS, filled with global_load_async_to_lds_b128
// (ASYNCcnt) for tile jt+1 while tile jt feeds the WMMAs from LDS.
// Attention + h' stores are non-temporal (written once, never re-read) so
// the 268 MB stream does not evict the WhT working set from L2.
// ---------------------------------------------------------------------------
__global__ void __launch_bounds__(256) k_main(const _Float16* __restrict__ WhT,
                                              const float* __restrict__ s1,
                                              const float* __restrict__ s2,
                                              const float* __restrict__ rowmax,
                                              const float* __restrict__ invden,
                                              float* __restrict__ h_out,
                                              float* __restrict__ att_out) {
    __shared__ __align__(64) _Float16 tile[2][DIM * 32];   // 2 x 16 KB

    const int tid  = threadIdx.x;
    const int lane = tid & 31;
    const int wave = tid >> 5;
    const int tl   = blockIdx.x * 8 + wave;   // 0..2047 (block never straddles a batch)
    const int row0 = tl * 16;
    const int b    = row0 / SEQ;
    const int m    = lane & 15;
    const int hi   = lane >> 4;
    const int r    = row0 + m;
    const int kb   = hi * 16;

    const float sv = s1[r];
    const float rm = rowmax[r];
    const float iv = invden[r];
    const float* s2b = s2 + b * SEQ;
    const _Float16* whb = WhT + (size_t)b * DIM * SEQ;
    float* attr = att_out + (size_t)r * SEQ;

    // Async-fill assignment: thread t copies row d = t of the tile (64 B = 4 x b128).
    const _Float16* grow = whb + (size_t)tid * SEQ;
    const unsigned ldsbase = (unsigned)(size_t)(&tile[0][0]);  // low 32 bits = LDS offset

    auto issue_fill = [&](int jt, int buf) {
        unsigned long long g = (unsigned long long)(size_t)(grow + jt * 32);
        unsigned l = ldsbase + (unsigned)buf * (DIM * 32 * 2) + (unsigned)tid * 64;
#pragma unroll
        for (int c = 0; c < 4; ++c) {
            asm volatile("global_load_async_to_lds_b128 %0, %1, off"
                         :: "v"(l + c * 16), "v"(g + c * 16)
                         : "memory");
        }
    };

    issue_fill(0, 0);
    asm volatile("s_wait_asynccnt 0x0" ::: "memory");
    __syncthreads();

    v8f acc[16] = {};

    for (int jt = 0; jt < 64; ++jt) {
        const int cur = jt & 1;
        if (jt < 63) issue_fill(jt + 1, cur ^ 1);   // overlap fill with compute

        const int j0 = jt * 32 + hi * 16;
        v8f z0 = *(const v8f*)(s2b + j0);          // j = j0 + e
        v8f z1 = *(const v8f*)(s2b + j0 + 8);      // j = j0 + 8 + e
        v8f a0, a1;
#pragma unroll
        for (int e = 0; e < 8; ++e) {
            float za = sv + z0[e];
            float la = za >= 0.f ? za : LEAKY * za;
            a0[e] = __expf(la - rm) * iv;
            float zb = sv + z1[e];
            float lb = zb >= 0.f ? zb : LEAKY * zb;
            a1[e] = __expf(lb - rm) * iv;
        }
        // Write attention (f32) exactly once, non-temporal streaming stores.
        __builtin_nontemporal_store(a0, (v8f*)(attr + j0));
        __builtin_nontemporal_store(a1, (v8f*)(attr + j0 + 8));

        // Pack WMMA A-tile (16x32 f16): elems 0-7 -> K=hi*16+e, 8-15 -> K=hi*16+8+e.
        v16h A;
#pragma unroll
        for (int e = 0; e < 8; ++e) { A[e] = (_Float16)a0[e]; A[e + 8] = (_Float16)a1[e]; }

        // B from LDS: tile[cur][d*32 + kb], d = nt*16 + m; 1-deep prefetch so the
        // ds_load for nt+1 overlaps the WMMA for nt.
        const _Float16* lb = &tile[cur][m * 32 + kb];
        v16h bm = *(const v16h*)lb;
#pragma unroll
        for (int nt = 0; nt < 16; ++nt) {
            v16h bn = (nt < 15) ? *(const v16h*)(lb + (nt + 1) * (16 * 32)) : bm;
            acc[nt] = __builtin_amdgcn_wmma_f32_16x16x32_f16(
                false, A, false, bm, (short)0, acc[nt], false, false);
            bm = bn;
        }

        // Drain the fill for jt+1, then block-wide barrier: after this, buf cur^1
        // is fully written and everyone is done reading buf cur.
        asm volatile("s_wait_asynccnt 0x0" ::: "memory");
        __syncthreads();
    }

    // Store h' (f32, row-major [b][i][d]) -- streaming, non-temporal.
#pragma unroll
    for (int nt = 0; nt < 16; ++nt) {
#pragma unroll
        for (int v = 0; v < 8; ++v) {
            int mi = v + 8 * hi;
            __builtin_nontemporal_store(acc[nt][v],
                &h_out[(size_t)(row0 + mi) * DIM + nt * 16 + m]);
        }
    }
}

// ---------------------------------------------------------------------------
// Launch
// ---------------------------------------------------------------------------
extern "C" void kernel_launch(void* const* d_in, const int* in_sizes, int n_in,
                              void* d_out, int out_size, void* d_ws, size_t ws_size,
                              hipStream_t stream) {
    const float* x = (const float*)d_in[0];   // (16, 2048, 256)
    const float* W = (const float*)d_in[1];   // (256, 256)
    const float* a = (const float*)d_in[2];   // (512, 1)

    char* ws = (char*)d_ws;
    _Float16* W16T = (_Float16*)ws;  ws += (size_t)DIM * DIM * 2;              // 128 KB
    _Float16* WhT  = (_Float16*)ws;  ws += (size_t)BATCH * DIM * SEQ * 2;      // 16 MB
    float* s1      = (float*)ws;     ws += (size_t)BATCH * SEQ * 4;
    float* s2      = (float*)ws;     ws += (size_t)BATCH * SEQ * 4;
    float* m2      = (float*)ws;     ws += 64;
    float* rmax    = (float*)ws;     ws += (size_t)BATCH * SEQ * 4;
    float* invden  = (float*)ws;     ws += (size_t)BATCH * SEQ * 4;

    float* h_out   = (float*)d_out;                                   // (16,2048,256)
    float* att_out = h_out + (size_t)BATCH * SEQ * DIM;               // (16,2048,2048)

    k_prep <<<256, 256, 0, stream>>>(W, W16T);
    k_gemm1<<<256, 256, 0, stream>>>(x, W16T, WhT);
    k_s12  <<<128, 256, 0, stream>>>(WhT, a, s1, s2);
    k_m2   <<< 16, 256, 0, stream>>>(s2, m2);
    k_denom<<<128, 256, 0, stream>>>(s1, s2, m2, rmax, invden);
    k_main <<<256, 256, 0, stream>>>(WhT, s1, s2, rmax, invden, h_out, att_out);
}
